// CEBlock_6760278524037
// MI455X (gfx1250) — matrix operand order repaired and checked by reading, not compiled, not run on hardware
//
#include <hip/hip_runtime.h>
#include <hip/hip_bf16.h>
#include <math.h>

// ---------------- problem constants ----------------
#define B_      32
#define LT_     64
#define LS_     256
#define L_      320      // LT+LS
#define DIM_    768
#define HEADS_  12
#define HD_     64
#define HID_    3072
#define KEEP_   180      // ceil(0.7*256)
#define REM_    76
#define LK_     244      // LT + KEEP
#define MROWS_  (B_*L_)   // 10240
#define PROWS_  (B_*LK_)  // 7808

// output flat offsets (elements)
#define OUT_X     0
#define OUT_GIT   (B_*LK_*DIM_)                 // 5996544
#define OUT_KEEP  (OUT_GIT + B_*LT_)            // +2048
#define OUT_REM   (OUT_KEEP + B_*KEEP_)         // +5760
#define OUT_ATTN  (OUT_REM + B_*REM_)           // +2432

typedef __attribute__((ext_vector_type(16))) _Float16 v16h;
typedef __attribute__((ext_vector_type(8)))  _Float16 v8h;
typedef __attribute__((ext_vector_type(8)))  float    v8f;

// ---------------- WMMA fragment helpers ----------------
// A fragment (16x32 f16, M x K): lane r=lane%16 is row; lanes<16 hold K{0..7,16..23},
// lanes>=16 hold K{8..15,24..31}. rowptr points at A[row] + k0.
__device__ __forceinline__ v16h load_fragA(const _Float16* rowptr, int lane) {
    int koff = (lane >> 4) * 8;
    v8h lo = *(const v8h*)(rowptr + koff);
    v8h hi = *(const v8h*)(rowptr + koff + 16);
    v16h f;
#pragma unroll
    for (int i = 0; i < 8; ++i) { f[i] = lo[i]; f[8 + i] = hi[i]; }
    return f;
}

// B fragment (32x16 f16, K x N) with operand stored K-contiguous per column
// (i.e. W[n][k] row-major along k). lane n=lane%16 is column; lanes<16 hold
// K{0..7,16..23}, lanes>=16 hold K{8..15,24..31}. rowptr points at W[col] + k0.
__device__ __forceinline__ v16h load_fragB(const _Float16* rowptr, int lane) {
    int kb = (lane >> 4) * 8;
    v8h lo = *(const v8h*)(rowptr + kb);
    v8h hi = *(const v8h*)(rowptr + kb + 16);
    v16h f;
#pragma unroll
    for (int i = 0; i < 8; ++i) { f[i] = lo[i]; f[8 + i] = hi[i]; }
    return f;
}

__device__ __forceinline__ v8f wmma_f16(v16h a, v16h b, v8f c) {
    return __builtin_amdgcn_wmma_f32_16x16x32_f16(false, a, false, b, (short)0, c, false, false);
}

__device__ __forceinline__ float gelu_exact(float v) {
    return 0.5f * v * (1.0f + erff(v * 0.70710678118654752f));
}

// ---------------- generic WMMA GEMM, register-blocked ----------------
// C[m][n] = epilogue( sum_k A16[m*lda+k] * W16[n*ldw+k] )
// Each wave computes a 32(M) x 64(N) macro-tile: 8 WMMAs per 32-wide k-step
// against 2 A-fragments + 4 B-fragments -> 21.3 FLOP/operand-byte.
// Block = 4 waves stacked in M -> 128 x 64 block tile.
// grid = (M/128, N/64), block = 128.
template<int ACT /*0 none, 1 gelu*/, bool HAS_BIAS, bool HAS_RES, bool OUT_F32, bool OUT_F16>
__global__ __launch_bounds__(128) void gemm_wmma_kernel(
    const _Float16* __restrict__ A, int lda,
    const _Float16* __restrict__ W, int ldw,
    const float* __restrict__ bias,
    const float* __restrict__ res, int ldr,
    float* __restrict__ Cf, _Float16* __restrict__ Ch, int ldc,
    int K)
{
    const int lane = threadIdx.x & 31;
    const int wave = threadIdx.x >> 5;
    const int r    = lane & 15;
    const int mbase = blockIdx.x * 128 + wave * 32;
    const int nbase = blockIdx.y * 64;

    const _Float16* arow0 = A + (size_t)(mbase + r)      * lda;
    const _Float16* arow1 = A + (size_t)(mbase + 16 + r) * lda;
    const _Float16* brow0 = W + (size_t)(nbase + r)      * ldw;
    const _Float16* brow1 = W + (size_t)(nbase + 16 + r) * ldw;
    const _Float16* brow2 = W + (size_t)(nbase + 32 + r) * ldw;
    const _Float16* brow3 = W + (size_t)(nbase + 48 + r) * ldw;

    v8f acc[2][4];
#pragma unroll
    for (int mi = 0; mi < 2; ++mi)
#pragma unroll
        for (int ni = 0; ni < 4; ++ni) acc[mi][ni] = (v8f){};

    for (int k0 = 0; k0 < K; k0 += 32) {
        // pull the next k-steps toward the WGP while this one computes
        __builtin_prefetch(arow0 + k0 + 128, 0, 3);
        __builtin_prefetch(brow0 + k0 + 128, 0, 3);

        v16h a0 = load_fragA(arow0 + k0, lane);
        v16h a1 = load_fragA(arow1 + k0, lane);
        v16h b0 = load_fragB(brow0 + k0, lane);
        v16h b1 = load_fragB(brow1 + k0, lane);
        v16h b2 = load_fragB(brow2 + k0, lane);
        v16h b3 = load_fragB(brow3 + k0, lane);

        acc[0][0] = wmma_f16(a0, b0, acc[0][0]);
        acc[0][1] = wmma_f16(a0, b1, acc[0][1]);
        acc[0][2] = wmma_f16(a0, b2, acc[0][2]);
        acc[0][3] = wmma_f16(a0, b3, acc[0][3]);
        acc[1][0] = wmma_f16(a1, b0, acc[1][0]);
        acc[1][1] = wmma_f16(a1, b1, acc[1][1]);
        acc[1][2] = wmma_f16(a1, b2, acc[1][2]);
        acc[1][3] = wmma_f16(a1, b3, acc[1][3]);
    }

    const int rhalf = (lane >> 4) * 8;
#pragma unroll
    for (int mi = 0; mi < 2; ++mi) {
#pragma unroll
        for (int ni = 0; ni < 4; ++ni) {
            const int col = nbase + ni * 16 + r;
            const float bval = HAS_BIAS ? bias[col] : 0.0f;
#pragma unroll
            for (int i = 0; i < 8; ++i) {
                int row = mbase + mi * 16 + rhalf + i;
                float v = acc[mi][ni][i];
                if (HAS_BIAS) v += bval;
                if (ACT == 1) v = gelu_exact(v);
                if (HAS_RES)  v += res[(size_t)row * ldr + col];
                if (OUT_F32)  Cf[(size_t)row * ldc + col] = v;
                if (OUT_F16)  Ch[(size_t)row * ldc + col] = (_Float16)v;
            }
        }
    }
}

// ---------------- layernorm (fp32 in, f16 out) ----------------
__global__ __launch_bounds__(256) void ln_kernel(const float* __restrict__ x,
                                                 const float* __restrict__ w,
                                                 const float* __restrict__ b,
                                                 _Float16* __restrict__ out16,
                                                 int /*rows via grid*/)
{
    __shared__ float red[256];
    const int row = blockIdx.x;
    const int tid = threadIdx.x;
    const float* xr = x + (size_t)row * DIM_;

    float s = 0.f;
    for (int j = tid; j < DIM_; j += 256) s += xr[j];
    red[tid] = s; __syncthreads();
    for (int st = 128; st > 0; st >>= 1) { if (tid < st) red[tid] += red[tid + st]; __syncthreads(); }
    float mu = red[0] * (1.0f / DIM_);
    __syncthreads();

    float v = 0.f;
    for (int j = tid; j < DIM_; j += 256) { float d = xr[j] - mu; v += d * d; }
    red[tid] = v; __syncthreads();
    for (int st = 128; st > 0; st >>= 1) { if (tid < st) red[tid] += red[tid + st]; __syncthreads(); }
    float rstd = rsqrtf(red[0] * (1.0f / DIM_) + 1e-5f);
    __syncthreads();

    for (int j = tid; j < DIM_; j += 256)
        out16[(size_t)row * DIM_ + j] = (_Float16)((xr[j] - mu) * rstd * w[j] + b[j]);
}

// ---------------- fp32 -> f16 convert ----------------
__global__ void cvt16_kernel(const float* __restrict__ in, _Float16* __restrict__ out, size_t n) {
    size_t i = (size_t)blockIdx.x * 256 + threadIdx.x;
    if (i < n) out[i] = (_Float16)in[i];
}

// ---------------- V transpose: vT[bh][d][t] = V[b][t][h][d] ----------------
__global__ void vtrans_kernel(const _Float16* __restrict__ qkv16, _Float16* __restrict__ vT16) {
    size_t idx = (size_t)blockIdx.x * 256 + threadIdx.x;
    const size_t total = (size_t)B_ * HEADS_ * HD_ * L_;
    if (idx >= total) return;
    int t  = (int)(idx % L_);
    int d  = (int)((idx / L_) % HD_);
    int bh = (int)(idx / ((size_t)L_ * HD_));
    int b = bh / HEADS_, h = bh % HEADS_;
    vT16[idx] = qkv16[(size_t)(b * L_ + t) * (3 * DIM_) + 2 * DIM_ + h * HD_ + d];
}

// ---------------- fused attention: S=QK^T*scale, softmax, attn out, O=P V ----------
// grid = (B*HEADS, L/16), block = 256 (8 waves)
__global__ __launch_bounds__(256) void attn_kernel(const _Float16* __restrict__ qkv16,
                                                   const _Float16* __restrict__ vT16,
                                                   float* __restrict__ attn_out,
                                                   _Float16* __restrict__ xattn16)
{
    __shared__ float    S[16][L_];   // 20 KB
    __shared__ _Float16 P[16][L_];   // 10 KB
    const int bh = blockIdx.x;
    const int b = bh / HEADS_, h = bh % HEADS_;
    const int mt = blockIdx.y;
    const int tid = threadIdx.x;
    const int wave = tid >> 5, lane = tid & 31;
    const int r = lane & 15;
    const int rhalf = (lane >> 4) * 8;
    const float scale = 0.125f;  // 1/sqrt(64)

    const _Float16* qrow = qkv16 + (size_t)(b * L_ + mt * 16 + r) * (3 * DIM_) + h * HD_;

    // phase 1: S tiles via WMMA
    for (int nt = wave; nt < L_ / 16; nt += 8) {
        const _Float16* krow = qkv16 + (size_t)(b * L_ + nt * 16 + r) * (3 * DIM_) + DIM_ + h * HD_;
        v8f acc = {};
#pragma unroll
        for (int k0 = 0; k0 < HD_; k0 += 32) {
            v16h af = load_fragA(qrow + k0, lane);
            v16h bf = load_fragB(krow + k0, lane);
            acc = wmma_f16(af, bf, acc);
        }
#pragma unroll
        for (int i = 0; i < 8; ++i)
            S[rhalf + i][nt * 16 + r] = acc[i] * scale;
    }
    __syncthreads();

    // phase 2: softmax, one row per thread 0..15
    if (tid < 16) {
        float mx = -3.0e38f;
        for (int j = 0; j < L_; ++j) mx = fmaxf(mx, S[tid][j]);
        float sum = 0.f;
        for (int j = 0; j < L_; ++j) { float e = __expf(S[tid][j] - mx); S[tid][j] = e; sum += e; }
        float inv = 1.0f / sum;
        for (int j = 0; j < L_; ++j) { float p = S[tid][j] * inv; S[tid][j] = p; P[tid][j] = (_Float16)p; }
    }
    __syncthreads();

    // write attn tensor (fp32 output)
    float* arow = attn_out + ((size_t)bh * L_ + mt * 16) * (size_t)L_;
    for (int idx = tid; idx < 16 * L_; idx += 256) {
        int rr = idx / L_, cc = idx % L_;
        arow[(size_t)rr * L_ + cc] = S[rr][cc];
    }

    // phase 3: O = P @ V via WMMA, waves 0..3 handle 16-column d-tiles
    if (wave < 4) {
        const _Float16* vrow = vT16 + ((size_t)bh * HD_ + wave * 16 + r) * (size_t)L_;
        const _Float16* prow = &P[r][0];
        v8f acc = {};
        for (int k0 = 0; k0 < L_; k0 += 32) {
            v16h af = load_fragA(prow + k0, lane);
            v16h bf = load_fragB(vrow + k0, lane);
            acc = wmma_f16(af, bf, acc);
        }
        int col = h * HD_ + wave * 16 + r;
#pragma unroll
        for (int i = 0; i < 8; ++i) {
            int row = mt * 16 + rhalf + i;
            xattn16[(size_t)(b * L_ + row) * DIM_ + col] = (_Float16)acc[i];
        }
    }
}

// ---------------- prune: scores, stable descending rank, indices -------------
__global__ __launch_bounds__(256) void prune_kernel(const float* __restrict__ attn,
                                                    const int* __restrict__ git,
                                                    const int* __restrict__ gis,
                                                    const unsigned char* __restrict__ mask,
                                                    int* __restrict__ keep_out,
                                                    int* __restrict__ rem_out,
                                                    int* __restrict__ git_out,
                                                    int* __restrict__ topk_ws)
{
    __shared__ float s[LS_];
    const int b = blockIdx.x;
    const int j = threadIdx.x;   // search token

    float msum = 0.f;
    for (int i = 0; i < LT_; ++i) msum += mask[b * LT_ + i] ? 1.f : 0.f;
    float acc = 0.f;
    for (int h = 0; h < HEADS_; ++h) {
        const float* base = attn + ((size_t)(b * HEADS_ + h) * L_) * (size_t)L_ + LT_ + j;
        float hs = 0.f;
        for (int i = 0; i < LT_; ++i)
            hs += mask[b * LT_ + i] ? base[(size_t)i * L_] : 0.f;
        acc += hs / msum;
    }
    s[j] = acc * (1.0f / HEADS_);
    __syncthreads();

    float mine = s[j];
    int rank = 0;
    for (int l = 0; l < LS_; ++l) {
        float o = s[l];
        rank += (o > mine) || (o == mine && l < j);
    }
    if (rank < KEEP_) {
        keep_out[b * KEEP_ + rank] = gis[b * LS_ + j];
        topk_ws[b * KEEP_ + rank]  = j;
    } else {
        rem_out[b * REM_ + (rank - KEEP_)] = gis[b * LS_ + j];
    }
    if (j < LT_) git_out[b * LT_ + j] = git[b * LT_ + j];
}

// ---------------- gather pruned tokens ----------------
__global__ __launch_bounds__(256) void gather_kernel(const float* __restrict__ x2,
                                                     const int* __restrict__ topk,
                                                     float* __restrict__ xp)
{
    const int row = blockIdx.x;            // 0..PROWS_-1
    const int b = row / LK_, rr = row % LK_;
    const int src = (rr < LT_) ? (b * L_ + rr) : (b * L_ + LT_ + topk[b * KEEP_ + (rr - LT_)]);
    const float* sp = x2 + (size_t)src * DIM_;
    float* dp = xp + (size_t)row * DIM_;
    for (int jj = threadIdx.x; jj < DIM_; jj += 256) dp[jj] = sp[jj];
}

// ---------------- host orchestration ----------------
static inline size_t align256(size_t x) { return (x + 255) & ~(size_t)255; }

extern "C" void kernel_launch(void* const* d_in, const int* in_sizes, int n_in,
                              void* d_out, int out_size, void* d_ws, size_t ws_size,
                              hipStream_t stream)
{
    (void)in_sizes; (void)n_in; (void)out_size; (void)ws_size;
    const float* x       = (const float*)d_in[0];
    const int*   git     = (const int*)d_in[1];
    const int*   gis     = (const int*)d_in[2];
    const unsigned char* mask = (const unsigned char*)d_in[3];
    const float* qkv_w   = (const float*)d_in[4];
    const float* proj_w  = (const float*)d_in[5];
    const float* proj_b  = (const float*)d_in[6];
    const float* n1w     = (const float*)d_in[7];
    const float* n1b     = (const float*)d_in[8];
    const float* n2w     = (const float*)d_in[9];
    const float* n2b     = (const float*)d_in[10];
    const float* fc1_w   = (const float*)d_in[11];
    const float* fc1_b   = (const float*)d_in[12];
    const float* fc2_w   = (const float*)d_in[13];
    const float* fc2_b   = (const float*)d_in[14];

    float* out_f   = (float*)d_out;
    int*   out_i   = (int*)d_out;
    float* attn_out = out_f + OUT_ATTN;

    // workspace carve-up
    char* wsp = (char*)d_ws;
    size_t off = 0;
    auto take = [&](size_t bytes) { char* p = wsp + off; off = align256(off + bytes); return p; };

    _Float16* h16     = (_Float16*)take((size_t)MROWS_ * DIM_ * 2);
    _Float16* qkvw16  = (_Float16*)take((size_t)3 * DIM_ * DIM_ * 2);
    _Float16* projw16 = (_Float16*)take((size_t)DIM_ * DIM_ * 2);
    _Float16* fc1w16  = (_Float16*)take((size_t)HID_ * DIM_ * 2);
    _Float16* fc2w16  = (_Float16*)take((size_t)DIM_ * HID_ * 2);
    _Float16* qkv16   = (_Float16*)take((size_t)MROWS_ * 3 * DIM_ * 2);
    _Float16* vT16    = (_Float16*)take((size_t)B_ * HEADS_ * HD_ * L_ * 2);
    _Float16* xattn16 = (_Float16*)take((size_t)MROWS_ * DIM_ * 2);
    _Float16* h2_16   = (_Float16*)take((size_t)PROWS_ * DIM_ * 2);
    _Float16* g16     = (_Float16*)take((size_t)PROWS_ * HID_ * 2);
    float*    x2      = (float*)take((size_t)MROWS_ * DIM_ * 4);
    float*    xp      = (float*)take((size_t)PROWS_ * DIM_ * 4);
    int*      topk    = (int*)take((size_t)B_ * KEEP_ * 4);

    // 1) weight conversions to f16
    {
        size_t n;
        n = (size_t)3 * DIM_ * DIM_;
        cvt16_kernel<<<(unsigned)((n + 255) / 256), 256, 0, stream>>>(qkv_w, qkvw16, n);
        n = (size_t)DIM_ * DIM_;
        cvt16_kernel<<<(unsigned)((n + 255) / 256), 256, 0, stream>>>(proj_w, projw16, n);
        n = (size_t)HID_ * DIM_;
        cvt16_kernel<<<(unsigned)((n + 255) / 256), 256, 0, stream>>>(fc1_w, fc1w16, n);
        n = (size_t)DIM_ * HID_;
        cvt16_kernel<<<(unsigned)((n + 255) / 256), 256, 0, stream>>>(fc2_w, fc2w16, n);
    }

    // 2) LN1 -> h16
    ln_kernel<<<MROWS_, 256, 0, stream>>>(x, n1w, n1b, h16, MROWS_);

    // 3) QKV GEMM: (10240 x 768) x (2304 x 768)^T -> qkv16 f16
    gemm_wmma_kernel<0, false, false, false, true>
        <<<dim3(MROWS_ / 128, (3 * DIM_) / 64), 128, 0, stream>>>(
            h16, DIM_, qkvw16, DIM_, nullptr, nullptr, 0,
            nullptr, qkv16, 3 * DIM_, DIM_);

    // 4) V transpose
    {
        size_t n = (size_t)B_ * HEADS_ * HD_ * L_;
        vtrans_kernel<<<(unsigned)((n + 255) / 256), 256, 0, stream>>>(qkv16, vT16);
    }

    // 5) attention (writes attn tensor to d_out and O tiles to xattn16)
    attn_kernel<<<dim3(B_ * HEADS_, L_ / 16), 256, 0, stream>>>(qkv16, vT16, attn_out, xattn16);

    // 6) proj GEMM + bias + residual(x) -> x2 fp32
    gemm_wmma_kernel<0, true, true, true, false>
        <<<dim3(MROWS_ / 128, DIM_ / 64), 128, 0, stream>>>(
            xattn16, DIM_, projw16, DIM_, proj_b, x, DIM_,
            x2, nullptr, DIM_, DIM_);

    // 7) prune: scores + stable argsort + index outputs
    prune_kernel<<<B_, 256, 0, stream>>>(attn_out, git, gis, mask,
                                         out_i + OUT_KEEP, out_i + OUT_REM,
                                         out_i + OUT_GIT, topk);

    // 8) gather pruned tokens -> xp fp32
    gather_kernel<<<PROWS_, 256, 0, stream>>>(x2, topk, xp);

    // 9) LN2 -> h2_16
    ln_kernel<<<PROWS_, 256, 0, stream>>>(xp, n2w, n2b, h2_16, PROWS_);

    // 10) FC1 GEMM + bias + exact GELU -> g16 f16
    gemm_wmma_kernel<1, true, false, false, true>
        <<<dim3(PROWS_ / 16 / 8, HID_ / 64), 128, 0, stream>>>(
            h2_16, DIM_, fc1w16, DIM_, fc1_b, nullptr, 0,
            nullptr, g16, HID_, DIM_);

    // 11) FC2 GEMM + bias + residual(xp) -> d_out x region fp32
    gemm_wmma_kernel<0, true, true, true, false>
        <<<dim3(PROWS_ / 16 / 8, DIM_ / 64), 128, 0, stream>>>(
            g16, HID_, fc2w16, HID_, fc2_b, xp, DIM_,
            out_f + OUT_X, nullptr, DIM_, HID_);
}
